// GCNConv_diag_2817498546211
// MI455X (gfx1250) — matrix operand order repaired and checked by reading, not compiled, not run on hardware
//
#include <hip/hip_runtime.h>
#include <hip/hip_bf16.h>

typedef __attribute__((ext_vector_type(16))) __bf16 v16bf;
typedef __attribute__((ext_vector_type(8)))  float  v8f;
typedef __attribute__((ext_vector_type(4)))  float  v4f;

#define N_ROWS 10000   // N (rows of A, rows of x)
#define F_COLS 256     // F (feature dim)
#define K_PAD  10016   // N rounded up to multiple of 32 (zero padded in Ht)

// ---------------------------------------------------------------------------
// Kernel 1: Ht[c][k] = bf16( x[k][c] * W[c] ), zero-padded for k in [10000,10016)
// Ht lives in d_ws: 256 * 10016 * 2 bytes = ~5.13 MB (L2-resident thereafter).
// ---------------------------------------------------------------------------
__global__ void __launch_bounds__(256)
scale_transpose_kernel(const float* __restrict__ x,
                       const float* __restrict__ W,
                       __bf16* __restrict__ Ht) {
    int idx = blockIdx.x * 256 + threadIdx.x;       // over 256*10016 = 2,564,096
    if (idx >= F_COLS * K_PAD) return;
    int c = idx / K_PAD;
    int k = idx - c * K_PAD;
    float v = 0.0f;
    if (k < N_ROWS) v = x[(long)k * F_COLS + c] * W[c];
    Ht[idx] = (__bf16)v;
}

// ---------------------------------------------------------------------------
// Kernel 2: out[16 x 256 panel per block] = A_panel @ Ht^T  via bf16 WMMA.
// 8 waves/block; wave w computes cols [w*32, w*32+32) as two 16x16 tiles.
// ---------------------------------------------------------------------------
__global__ void __launch_bounds__(256)
gemm_wmma_bf16_kernel(const float* __restrict__ A,
                      const __bf16* __restrict__ Ht,
                      float* __restrict__ out) {
    const int lane = threadIdx.x & 31;
    const int wave = threadIdx.x >> 5;            // 0..7
    const int m0   = blockIdx.x * 16;             // row-tile base
    const int n0   = wave * 32;                   // col base (two 16-wide tiles)
    const int l15  = lane & 15;
    const int hi   = (lane >> 4) & 1;             // half-wave select

    // A fragment addressing (16-bit A layout, ISA 7.12.2):
    //   lane l, M = l%16; VGPRs 0-3 hold K = kA_off+0..7, VGPRs 4-7 hold K = 16+kA_off+0..7
    //   where kA_off = 0 for lanes 0-15, 8 for lanes 16-31.
    const long arow   = (long)(m0 + l15) * N_ROWS;
    const int  kA_off = hi * 8;
    // B fragment addressing: lane l -> column n0(+16)+l%16, K = kB_off+0..7 and 16+kB_off+0..7, kB_off = 0/8 per half-wave.
    const int  kB_off = hi * 8;

    const __bf16* bt0 = Ht + (long)(n0 + l15)      * K_PAD;
    const __bf16* bt1 = Ht + (long)(n0 + 16 + l15) * K_PAD;

    v8f c0 = {};
    v8f c1 = {};

    int k0 = 0;
    #pragma unroll 2
    for (; k0 < 9984; k0 += 32) {
        const float* ap = A + arow + k0 + kA_off;
        // prefetch the A stream ~16 K-iterations ahead (global_prefetch_b8)
        __builtin_prefetch(ap + 512, 0, 3);

        v4f a0 = *(const v4f*)(ap);         // K = k0+kA_off   .. +3
        v4f a1 = *(const v4f*)(ap + 4);     // K = k0+kA_off+4 .. +7
        v4f a2 = *(const v4f*)(ap + 16);    // K = k0+kA_off+16.. +19
        v4f a3 = *(const v4f*)(ap + 20);    // K = k0+kA_off+20.. +23

        v16bf af;
        #pragma unroll
        for (int i = 0; i < 4; ++i) {
            af[i]      = (__bf16)a0[i];
            af[4 + i]  = (__bf16)a1[i];
            af[8 + i]  = (__bf16)a2[i];
            af[12 + i] = (__bf16)a3[i];
        }

        v16bf b0, b1;
        #pragma unroll
        for (int i = 0; i < 8; ++i) {
            b0[i]     = bt0[k0 + kB_off + i];
            b0[8 + i] = bt0[k0 + 16 + kB_off + i];
            b1[i]     = bt1[k0 + kB_off + i];
            b1[8 + i] = bt1[k0 + 16 + kB_off + i];
        }

        c0 = __builtin_amdgcn_wmma_f32_16x16x32_bf16(false, af, false, b0,
                                                     (short)0, c0, false, false);
        c1 = __builtin_amdgcn_wmma_f32_16x16x32_bf16(false, af, false, b1,
                                                     (short)0, c1, false, false);
    }

    // K tail: k0 == 9984, only 16 valid K values. Upper half of A fragment is
    // zero; B reads the zero-padded region of Ht.
    {
        const float* ap = A + arow + k0 + kA_off;
        v4f a0 = *(const v4f*)(ap);
        v4f a1 = *(const v4f*)(ap + 4);

        v16bf af = {};
        #pragma unroll
        for (int i = 0; i < 4; ++i) {
            af[i]     = (__bf16)a0[i];
            af[4 + i] = (__bf16)a1[i];
        }

        v16bf b0, b1;
        #pragma unroll
        for (int i = 0; i < 8; ++i) {
            b0[i]     = bt0[k0 + kB_off + i];
            b0[8 + i] = bt0[k0 + 16 + kB_off + i];
            b1[i]     = bt1[k0 + kB_off + i];
            b1[8 + i] = bt1[k0 + 16 + kB_off + i];
        }

        c0 = __builtin_amdgcn_wmma_f32_16x16x32_bf16(false, af, false, b0,
                                                     (short)0, c0, false, false);
        c1 = __builtin_amdgcn_wmma_f32_16x16x32_bf16(false, af, false, b1,
                                                     (short)0, c1, false, false);
    }

    // Store: C/D layout — VGPR r: lanes 0-15 -> M=r, lanes 16-31 -> M=r+8; N=lane%16.
    #pragma unroll
    for (int r = 0; r < 8; ++r) {
        const long row = (long)(m0 + r + hi * 8) * F_COLS;
        out[row + n0 + l15]      = c0[r];
        out[row + n0 + 16 + l15] = c1[r];
    }
}

// ---------------------------------------------------------------------------
extern "C" void kernel_launch(void* const* d_in, const int* in_sizes, int n_in,
                              void* d_out, int out_size, void* d_ws, size_t ws_size,
                              hipStream_t stream) {
    const float* x = (const float*)d_in[0];   // [10000, 256]
    const float* A = (const float*)d_in[1];   // [10000, 10000]
    const float* W = (const float*)d_in[2];   // [256]
    float* out = (float*)d_out;               // [10000, 256]
    __bf16* Ht = (__bf16*)d_ws;               // [256, 10016] bf16

    const int total = F_COLS * K_PAD;
    scale_transpose_kernel<<<(total + 255) / 256, 256, 0, stream>>>(x, W, Ht);
    gemm_wmma_bf16_kernel<<<N_ROWS / 16, 256, 0, stream>>>(A, Ht, out);
}
